// GNNCorrection_34849364640431
// MI455X (gfx1250) — matrix operand order repaired and checked by reading, not compiled, not run on hardware
//
#include <hip/hip_runtime.h>
#include <hip/hip_bf16.h>

// ---------------------------------------------------------------------------
// CDNA5 / gfx1250 GNN correction network.
// All GEMMs run on V_WMMA_F32_16X16X32_BF16 (wave32, 16x16 tiles, K-tiles of 32).
// Edge kernel: 128-thread block (4 waves) = 64 edges; 4 A-tiles per K-step,
//   N (128) split 2 n-tiles per wave -> each B fragment feeds 4 WMMAs.
// Node kernels: 64-thread block (2 waves) = 32 rows; each B fragment feeds 2.
// ---------------------------------------------------------------------------

typedef __attribute__((ext_vector_type(16))) __bf16 v16bf;
typedef __attribute__((ext_vector_type(8)))  float  v8f;

union FragAB {
    v16bf v;
    uint4 q[2];
};

__device__ __forceinline__ unsigned short f2bf(float f) {
    union { float f; unsigned u; } x;
    x.f = f;
    unsigned r = x.u + 0x7FFFu + ((x.u >> 16) & 1u);  // round-to-nearest-even
    return (unsigned short)(r >> 16);
}

__device__ __forceinline__ float silu_f(float x) {
    return x / (1.0f + __expf(-x));
}

__device__ __forceinline__ v8f wmma_bf16(const FragAB& a, const FragAB& b, v8f c) {
    return __builtin_amdgcn_wmma_f32_16x16x32_bf16(
        /*neg_a=*/false, a.v, /*neg_b=*/false, b.v,
        /*c_mod=*/(short)0, c, /*reuse_a=*/false, /*reuse_b=*/false);
}

// A-fragment (16x32 bf16 tile) from a row-major bf16 buffer.
// ISA layout: lane L holds row M=L%16; elements 0..7 -> K = k0 + (L/16)*8 + e,
// elements 8..15 -> K = k0 + 16 + (L/16)*8 + (e-8).  => two b128 loads.
__device__ __forceinline__ FragAB load_a(const unsigned short* base, int ld, int lane) {
    FragAB f;
    const unsigned short* p = base + (size_t)(lane & 15) * ld + (lane >> 4) * 8;
    f.q[0] = *(const uint4*)(p);
    f.q[1] = *(const uint4*)(p + 16);
    return f;
}

// B-fragment from pre-packed weights: frag-major, 32 lanes x 16 bf16 contiguous.
__device__ __forceinline__ FragAB load_b(const unsigned short* packed, int frag, int lane) {
    FragAB f;
    const uint4* p = (const uint4*)(packed + ((size_t)frag * 32 + lane) * 16);
    f.q[0] = p[0];
    f.q[1] = p[1];
    return f;
}

__device__ __forceinline__ void atom_add_f32(float* p, float v) {
    __hip_atomic_fetch_add(p, v, __ATOMIC_RELAXED, __HIP_MEMORY_SCOPE_AGENT);
}

// ---------------------------------------------------------------------------
// Pack a KxN fp32 weight matrix into WMMA B fragments (bf16), zero-padding K
// up to ktiles*32.  B layout: element e of lane L = W[kt*32 + (L/16)*8 + e + (e>=8 ? 8 : 0)]
// [nt*16 + L%16].
// ---------------------------------------------------------------------------
__global__ __launch_bounds__(32)
void pack_b_kernel(const float* __restrict__ W, int K, int N,
                   unsigned short* __restrict__ out) {
    int ntiles = N >> 4;
    int frag   = blockIdx.x;            // kt*ntiles + nt
    int kt     = frag / ntiles;
    int nt     = frag - kt * ntiles;
    int lane   = threadIdx.x;
    int n      = nt * 16 + (lane & 15);
    int kb     = kt * 32 + (lane >> 4) * 8;
    unsigned short* o = out + ((size_t)frag * 32 + lane) * 16;
#pragma unroll
    for (int e = 0; e < 16; ++e) {
        int k = kb + e + (e >= 8 ? 8 : 0);
        o[e] = f2bf((k < K) ? W[(size_t)k * N + n] : 0.0f);
    }
}

// ---------------------------------------------------------------------------
// x = concat(node_features[32], emb[species][128], coord[1]) @ in_W + in_b
// K = 161 padded to 192 (6 k-tiles).  32 nodes / block.  Writes x fp32 + bf16.
// ---------------------------------------------------------------------------
__global__ __launch_bounds__(64)
void node_input_kernel(const float* __restrict__ nf, const int* __restrict__ species,
                       const float* __restrict__ coord, const float* __restrict__ emb,
                       const unsigned short* __restrict__ Wp, const float* __restrict__ bias,
                       unsigned short* __restrict__ xb, float* __restrict__ xf) {
    __shared__ __align__(16) unsigned short A[32 * 192];
    int tid  = threadIdx.x;
    int w    = tid >> 5, lane = tid & 31;
    int n0   = blockIdx.x * 32;

    for (int i = tid; i < 32 * 192; i += 64) {
        int r = i / 192, c = i - r * 192;
        int node = n0 + r;
        float v;
        if (c < 32)        v = nf[(size_t)node * 32 + c];
        else if (c < 160)  v = emb[(size_t)species[node] * 128 + (c - 32)];
        else if (c == 160) v = coord[node];
        else               v = 0.0f;
        A[i] = f2bf(v);
    }
    __syncthreads();

    v8f acc[2][4] = {};
    for (int kt = 0; kt < 6; ++kt) {
        FragAB a0 = load_a(&A[kt * 32], 192, lane);
        FragAB a1 = load_a(&A[16 * 192 + kt * 32], 192, lane);
#pragma unroll
        for (int ntl = 0; ntl < 4; ++ntl) {
            FragAB b = load_b(Wp, kt * 8 + (w * 4 + ntl), lane);
            acc[0][ntl] = wmma_bf16(a0, b, acc[0][ntl]);
            acc[1][ntl] = wmma_bf16(a1, b, acc[1][ntl]);
        }
    }

    int lo = lane & 15, hi = lane >> 4;
#pragma unroll
    for (int t = 0; t < 2; ++t) {
#pragma unroll
        for (int ntl = 0; ntl < 4; ++ntl) {
            int n = (w * 4 + ntl) * 16 + lo;
            float bv = bias[n];
#pragma unroll
            for (int v = 0; v < 8; ++v) {
                int m = t * 16 + v + 8 * hi;  // D: elem v of lane L = D[v+8*(L/16)][L%16]
                float val = acc[t][ntl][v] + bv;
                size_t o = (size_t)(n0 + m) * 128 + n;
                xf[o] = val;
                xb[o] = f2bf(val);
            }
        }
    }
}

// ---------------------------------------------------------------------------
// Edge message MLP + scatter-add.  64 edges / block (4 waves, 128 threads).
// A row = [x[src](128) | x[dst](128) | ef(32) | c[src] | c[dst] | pad] K=290->320.
// msg = silu(A@mW1+mb1)@mW2+mb2 ; agg[dst] += msg (f32 atomics in L2).
// Wave w owns n-tiles {2w, 2w+1}; 4 row-tiles -> each B fragment feeds 4 WMMAs.
// ---------------------------------------------------------------------------
__global__ __launch_bounds__(128)
void edge_msg_kernel(const unsigned short* __restrict__ xb, const float* __restrict__ coord,
                     const int* __restrict__ ei, const float* __restrict__ ef,
                     const unsigned short* __restrict__ W1p, const float* __restrict__ b1,
                     const unsigned short* __restrict__ W2p, const float* __restrict__ b2,
                     float* __restrict__ agg, int E) {
    __shared__ __align__(16) unsigned short A[64 * 320];  // 40960 B; reused for h1 [64][128]
    int tid  = threadIdx.x;
    int w    = tid >> 5, lane = tid & 31;
    int e0   = blockIdx.x * 64;
    const int* src = ei;
    const int* dst = ei + E;

    // gather x[src] rows (64 rows x 16 uint4 chunks)
    for (int i = tid; i < 1024; i += 128) {
        int r = i >> 4, c = i & 15;
        int s = src[e0 + r];
        ((uint4*)&A[r * 320])[c] = ((const uint4*)(xb + (size_t)s * 128))[c];
    }
    // gather x[dst] rows
    for (int i = tid; i < 1024; i += 128) {
        int r = i >> 4, c = i & 15;
        int d = dst[e0 + r];
        ((uint4*)&A[r * 320 + 128])[c] = ((const uint4*)(xb + (size_t)d * 128))[c];
    }
    // edge features, coords, zero pad (cols 256..319)
    for (int i = tid; i < 64 * 64; i += 128) {
        int r = i >> 6, c = i & 63;
        float v;
        if (c < 32)       v = ef[(size_t)(e0 + r) * 32 + c];
        else if (c == 32) v = coord[src[e0 + r]];
        else if (c == 33) v = coord[dst[e0 + r]];
        else              v = 0.0f;
        A[r * 320 + 256 + c] = f2bf(v);
    }
    __syncthreads();

    v8f acc[4][2] = {};
    for (int kt = 0; kt < 10; ++kt) {
        FragAB a0 = load_a(&A[kt * 32], 320, lane);
        FragAB a1 = load_a(&A[16 * 320 + kt * 32], 320, lane);
        FragAB a2 = load_a(&A[32 * 320 + kt * 32], 320, lane);
        FragAB a3 = load_a(&A[48 * 320 + kt * 32], 320, lane);
#pragma unroll
        for (int ntl = 0; ntl < 2; ++ntl) {
            FragAB b = load_b(W1p, kt * 8 + (w * 2 + ntl), lane);
            acc[0][ntl] = wmma_bf16(a0, b, acc[0][ntl]);
            acc[1][ntl] = wmma_bf16(a1, b, acc[1][ntl]);
            acc[2][ntl] = wmma_bf16(a2, b, acc[2][ntl]);
            acc[3][ntl] = wmma_bf16(a3, b, acc[3][ntl]);
        }
    }
    __syncthreads();   // all waves done reading A before h1 overwrites it

    // silu -> h1 (bf16) into LDS, ld = 128 (reuse A region)
    int lo = lane & 15, hi = lane >> 4;
#pragma unroll
    for (int t = 0; t < 4; ++t) {
#pragma unroll
        for (int ntl = 0; ntl < 2; ++ntl) {
            int n = (w * 2 + ntl) * 16 + lo;
            float bv = b1[n];
#pragma unroll
            for (int v = 0; v < 8; ++v) {
                int m = t * 16 + v + 8 * hi;
                A[m * 128 + n] = f2bf(silu_f(acc[t][ntl][v] + bv));
            }
        }
    }
    __syncthreads();

    v8f acc2[4][2] = {};
    for (int kt = 0; kt < 4; ++kt) {
        FragAB a0 = load_a(&A[kt * 32], 128, lane);
        FragAB a1 = load_a(&A[16 * 128 + kt * 32], 128, lane);
        FragAB a2 = load_a(&A[32 * 128 + kt * 32], 128, lane);
        FragAB a3 = load_a(&A[48 * 128 + kt * 32], 128, lane);
#pragma unroll
        for (int ntl = 0; ntl < 2; ++ntl) {
            FragAB b = load_b(W2p, kt * 8 + (w * 2 + ntl), lane);
            acc2[0][ntl] = wmma_bf16(a0, b, acc2[0][ntl]);
            acc2[1][ntl] = wmma_bf16(a1, b, acc2[1][ntl]);
            acc2[2][ntl] = wmma_bf16(a2, b, acc2[2][ntl]);
            acc2[3][ntl] = wmma_bf16(a3, b, acc2[3][ntl]);
        }
    }

#pragma unroll
    for (int t = 0; t < 4; ++t) {
#pragma unroll
        for (int ntl = 0; ntl < 2; ++ntl) {
            int n = (w * 2 + ntl) * 16 + lo;
            float bv = b2[n];
#pragma unroll
            for (int v = 0; v < 8; ++v) {
                int m = t * 16 + v + 8 * hi;
                int d = dst[e0 + m];
                atom_add_f32(&agg[(size_t)d * 128 + n], acc2[t][ntl][v] + bv);
            }
        }
    }
}

// ---------------------------------------------------------------------------
// Node update MLP + residual + LayerNorm (fused).  K = 257 -> 288 (9 k-tiles).
// 32 nodes / block.  x <- LN(x + silu(concat(x,agg,coord)@uW1+ub1)@uW2+ub2).
// ---------------------------------------------------------------------------
__global__ __launch_bounds__(64)
void node_update_kernel(unsigned short* __restrict__ xb, float* __restrict__ xf,
                        const float* __restrict__ agg, const float* __restrict__ coord,
                        const unsigned short* __restrict__ W1p, const float* __restrict__ b1,
                        const unsigned short* __restrict__ W2p, const float* __restrict__ b2,
                        const float* __restrict__ ln_g, const float* __restrict__ ln_b) {
    __shared__ __align__(16) unsigned short A[32 * 288];  // 18432 B; reused as f32[32][128]
    int tid  = threadIdx.x;
    int w    = tid >> 5, lane = tid & 31;
    int n0   = blockIdx.x * 32;

    // cols 0..127: x (bf16 copy, vectorized)
    for (int i = tid; i < 512; i += 64) {
        int r = i >> 4, c = i & 15;
        ((uint4*)&A[r * 288])[c] = ((const uint4*)(xb + (size_t)(n0 + r) * 128))[c];
    }
    // cols 128..287: agg, coord, pad
    for (int i = tid; i < 32 * 160; i += 64) {
        int r = i / 160, c = i - r * 160;
        float v;
        if (c < 128)       v = agg[(size_t)(n0 + r) * 128 + c];
        else if (c == 128) v = coord[n0 + r];
        else               v = 0.0f;
        A[r * 288 + 128 + c] = f2bf(v);
    }
    __syncthreads();

    v8f acc[2][4] = {};
    for (int kt = 0; kt < 9; ++kt) {
        FragAB a0 = load_a(&A[kt * 32], 288, lane);
        FragAB a1 = load_a(&A[16 * 288 + kt * 32], 288, lane);
#pragma unroll
        for (int ntl = 0; ntl < 4; ++ntl) {
            FragAB b = load_b(W1p, kt * 8 + (w * 4 + ntl), lane);
            acc[0][ntl] = wmma_bf16(a0, b, acc[0][ntl]);
            acc[1][ntl] = wmma_bf16(a1, b, acc[1][ntl]);
        }
    }
    __syncthreads();

    int lo = lane & 15, hi = lane >> 4;
#pragma unroll
    for (int t = 0; t < 2; ++t) {
#pragma unroll
        for (int ntl = 0; ntl < 4; ++ntl) {
            int n = (w * 4 + ntl) * 16 + lo;
            float bv = b1[n];
#pragma unroll
            for (int v = 0; v < 8; ++v) {
                int m = t * 16 + v + 8 * hi;
                A[m * 128 + n] = f2bf(silu_f(acc[t][ntl][v] + bv));
            }
        }
    }
    __syncthreads();

    v8f acc2[2][4] = {};
    for (int kt = 0; kt < 4; ++kt) {
        FragAB a0 = load_a(&A[kt * 32], 128, lane);
        FragAB a1 = load_a(&A[16 * 128 + kt * 32], 128, lane);
#pragma unroll
        for (int ntl = 0; ntl < 4; ++ntl) {
            FragAB b = load_b(W2p, kt * 8 + (w * 4 + ntl), lane);
            acc2[0][ntl] = wmma_bf16(a0, b, acc2[0][ntl]);
            acc2[1][ntl] = wmma_bf16(a1, b, acc2[1][ntl]);
        }
    }
    __syncthreads();

    // y = x + upd into LDS as f32 [32][128]
    float* Y = (float*)A;
#pragma unroll
    for (int t = 0; t < 2; ++t) {
#pragma unroll
        for (int ntl = 0; ntl < 4; ++ntl) {
            int n = (w * 4 + ntl) * 16 + lo;
            float bv = b2[n];
#pragma unroll
            for (int v = 0; v < 8; ++v) {
                int m = t * 16 + v + 8 * hi;
                Y[m * 128 + n] = xf[(size_t)(n0 + m) * 128 + n] + acc2[t][ntl][v] + bv;
            }
        }
    }
    __syncthreads();

    // LayerNorm: one row per thread (32 rows, threads 0..31)
    if (tid < 32) {
        int node = n0 + tid;
        const float* y = Y + tid * 128;
        float mu = 0.0f;
        for (int n = 0; n < 128; ++n) mu += y[n];
        mu *= (1.0f / 128.0f);
        float var = 0.0f;
        for (int n = 0; n < 128; ++n) { float d = y[n] - mu; var += d * d; }
        var *= (1.0f / 128.0f);
        float rs = rsqrtf(var + 1e-5f);
        for (int n = 0; n < 128; ++n) {
            float o = (y[n] - mu) * rs * ln_g[n] + ln_b[n];
            size_t idx = (size_t)node * 128 + n;
            xf[idx] = o;
            xb[idx] = f2bf(o);
        }
    }
}

// ---------------------------------------------------------------------------
// Readout: out = silu(x @ rW1 + rb1) @ rW2 + rb2.  rW1: 128x64 via WMMA.
// 32 nodes / block; n-tiles (4) split 2/2 across the waves.
// ---------------------------------------------------------------------------
__global__ __launch_bounds__(64)
void readout_kernel(const unsigned short* __restrict__ xb,
                    const unsigned short* __restrict__ W1p, const float* __restrict__ rb1,
                    const float* __restrict__ rW2, const float* __restrict__ rb2,
                    float* __restrict__ out) {
    __shared__ __align__(16) float H[32 * 64];
    int tid  = threadIdx.x;
    int w    = tid >> 5, lane = tid & 31;
    int n0   = blockIdx.x * 32;

    v8f acc[2][2] = {};
    for (int kt = 0; kt < 4; ++kt) {
        FragAB a0 = load_a(xb + (size_t)n0 * 128 + kt * 32, 128, lane);        // rows 0-15
        FragAB a1 = load_a(xb + (size_t)(n0 + 16) * 128 + kt * 32, 128, lane); // rows 16-31
#pragma unroll
        for (int ntl = 0; ntl < 2; ++ntl) {
            FragAB b = load_b(W1p, kt * 4 + (w * 2 + ntl), lane);
            acc[0][ntl] = wmma_bf16(a0, b, acc[0][ntl]);
            acc[1][ntl] = wmma_bf16(a1, b, acc[1][ntl]);
        }
    }

    int lo = lane & 15, hi = lane >> 4;
#pragma unroll
    for (int t = 0; t < 2; ++t) {
#pragma unroll
        for (int ntl = 0; ntl < 2; ++ntl) {
            int n = (w * 2 + ntl) * 16 + lo;
            float bv = rb1[n];
#pragma unroll
            for (int v = 0; v < 8; ++v) {
                int m = t * 16 + v + 8 * hi;
                H[m * 64 + n] = silu_f(acc[t][ntl][v] + bv);
            }
        }
    }
    __syncthreads();

    if (tid < 32) {
        float s = rb2[0];
        for (int j = 0; j < 64; ++j) s += H[tid * 64 + j] * rW2[j];
        out[n0 + tid] = s;
    }
}

// ---------------------------------------------------------------------------
// Host-side launch
// ---------------------------------------------------------------------------
extern "C" void kernel_launch(void* const* d_in, const int* in_sizes, int n_in,
                              void* d_out, int out_size, void* d_ws, size_t ws_size,
                              hipStream_t stream) {
    const float* nf      = (const float*)d_in[0];
    const int*   species = (const int*)d_in[1];
    const int*   ei      = (const int*)d_in[2];
    const float* ef      = (const float*)d_in[3];
    const float* coord   = (const float*)d_in[4];
    int N = in_sizes[1];        // 40000
    int E = in_sizes[3] / 32;   // 640000

    // --- classify flattened params by element count (JAX sorted-key tree order
    //     for same-size leaves: in_W, in_b, [b,g,mW1,mW2,mb1,mb2,uW1,uW2,ub1,ub2]x2,
    //     rW1, rW2, rb1, rb2, species_emb) ---
    const float *in_W = nullptr, *emb = nullptr, *rW1 = nullptr, *rb2 = nullptr;
    const float *mW1[2] = {}, *uW1[2] = {}, *w16[4] = {};
    const float *v128[16] = {}; const float *v64[2] = {};
    int nmW1 = 0, nuW1 = 0, n16 = 0, n128 = 0, n64 = 0;
    for (int i = 5; i < n_in; ++i) {
        const float* p = (const float*)d_in[i];
        switch (in_sizes[i]) {
            case 20608: in_W = p; break;                       // 161x128
            case 12800: emb = p; break;                        // 100x128
            case 37120: if (nmW1 < 2) mW1[nmW1++] = p; break;  // 290x128
            case 32896: if (nuW1 < 2) uW1[nuW1++] = p; break;  // 257x128
            case 16384: if (n16 < 4) w16[n16++] = p; break;    // mW2_0,uW2_0,mW2_1,uW2_1
            case 8192:  rW1 = p; break;                        // 128x64
            case 128:   if (n128 < 16) v128[n128++] = p; break;
            case 64:    if (n64 < 2) v64[n64++] = p; break;    // rW2, rb1
            case 1:     rb2 = p; break;
            default: break;
        }
    }
    const float* in_b   = v128[0];
    const float* mW2[2] = { w16[0], w16[2] };
    const float* uW2[2] = { w16[1], w16[3] };
    const float *lb[2], *lg[2], *mb1[2], *mb2[2], *ub1[2], *ub2[2];
    for (int l = 0; l < 2; ++l) {
        const float* const* base = v128 + 1 + l * 6;  // b, g, mb1, mb2, ub1, ub2
        lb[l] = base[0]; lg[l] = base[1]; mb1[l] = base[2];
        mb2[l] = base[3]; ub1[l] = base[4]; ub2[l] = base[5];
    }
    const float* rW2 = v64[0];
    const float* rb1 = v64[1];

    // --- workspace carve (~52 MB) ---
    char* ws = (char*)d_ws;
    size_t off = 0;
    auto take = [&](size_t bytes) -> void* {
        void* p = ws + off;
        off = (off + bytes + 255) & ~(size_t)255;
        return p;
    };
    unsigned short* xb  = (unsigned short*)take((size_t)N * 128 * 2);
    float*          xf  = (float*)take((size_t)N * 128 * 4);
    float*          agg = (float*)take((size_t)N * 128 * 4);
    auto takeW = [&](int frags) -> unsigned short* {
        return (unsigned short*)take((size_t)frags * 512 * 2);  // 32 lanes x 16 bf16 per frag
    };
    unsigned short* inWp    = takeW(6 * 8);
    unsigned short* mW1p[2] = { takeW(80), takeW(80) };
    unsigned short* mW2p[2] = { takeW(32), takeW(32) };
    unsigned short* uW1p[2] = { takeW(72), takeW(72) };
    unsigned short* uW2p[2] = { takeW(32), takeW(32) };
    unsigned short* rW1p    = takeW(16);
    (void)ws_size; (void)out_size;

    // --- pack weights to WMMA B-fragment layout (cheap; done every call) ---
    pack_b_kernel<<<48, 32, 0, stream>>>(in_W, 161, 128, inWp);
    for (int l = 0; l < 2; ++l) {
        pack_b_kernel<<<80, 32, 0, stream>>>(mW1[l], 290, 128, mW1p[l]);
        pack_b_kernel<<<32, 32, 0, stream>>>(mW2[l], 128, 128, mW2p[l]);
        pack_b_kernel<<<72, 32, 0, stream>>>(uW1[l], 257, 128, uW1p[l]);
        pack_b_kernel<<<32, 32, 0, stream>>>(uW2[l], 128, 128, uW2p[l]);
    }
    pack_b_kernel<<<16, 32, 0, stream>>>(rW1, 128, 64, rW1p);

    // --- network ---
    node_input_kernel<<<N / 32, 64, 0, stream>>>(nf, species, coord, emb, inWp, in_b, xb, xf);
    for (int l = 0; l < 2; ++l) {
        hipMemsetAsync(agg, 0, (size_t)N * 128 * 4, stream);
        edge_msg_kernel<<<E / 64, 128, 0, stream>>>(xb, coord, ei, ef, mW1p[l], mb1[l],
                                                    mW2p[l], mb2[l], agg, E);
        node_update_kernel<<<N / 32, 64, 0, stream>>>(xb, xf, agg, coord, uW1p[l], ub1[l],
                                                      uW2p[l], ub2[l], lg[l], lb[l]);
    }
    readout_kernel<<<N / 32, 64, 0, stream>>>(xb, rW1p, rb1, rW2, rb2, (float*)d_out);
}
